// NeuronMLPBlock_23819888623798
// MI455X (gfx1250) — matrix operand order repaired and checked by reading, not compiled, not run on hardware
//
#include <hip/hip_runtime.h>
#include <hip/hip_bf16.h>
#include <math.h>

// ---------------- problem constants (from reference) ----------------
#define ALPHA_  1.702f
#define LIMIT_  7.0f
#define EPS_    1e-5f

#define E_      8
#define TOPK_   2
#define H_      2048
#define I_      2048
#define TWO_I   4096
#define T_      1024
#define KSTEP   32
#define KT_H    (H_ / KSTEP)   // 64 k-steps over H
#define KT_I    (I_ / KSTEP)   // 64 k-steps over I

// GEMM tiling: block = 256 threads = 8 waves
// each wave: 4 M-tiles x 4 N-tiles of 16x16 -> 64 rows x 64 cols
// block: M = 64 rows, N = 8 waves * 64 = 512 cols
#define MTILE   64
#define NBLK    512

// CDNA5 async global->LDS path (guarded: builtin verified present by probe round)
#if defined(__gfx1250__) && __has_builtin(__builtin_amdgcn_global_load_async_to_lds_b128)
#define HAVE_ASYNC_LDS 1
#endif

// ---------------- workspace layout (bytes) ----------------
#define OFF_TBF   ((size_t)0)                       // T*H bf16       = 4 MB
#define OFF_GATES (OFF_TBF   + (size_t)T_*H_*2)     // E*T float      = 32 KB
#define OFF_TOK   (OFF_GATES + (size_t)E_*T_*4)     // E*T int        = 32 KB
#define OFF_CNT   (OFF_TOK   + (size_t)E_*T_*4)     // E ints (pad)
#define OFF_ACT   (OFF_CNT   + (size_t)1024)        // E*T*I bf16     = 32 MB

// ---------------- vector types for WMMA ----------------
typedef __attribute__((ext_vector_type(16))) __bf16       v16bf;
typedef __attribute__((ext_vector_type(8)))  float        v8f;
typedef __attribute__((ext_vector_type(4)))  unsigned int u32x4;

// GCC-style int4 vector matching the async builtin's expected pointee type
typedef int v4i_g __attribute__((vector_size(16)));
#ifdef HAVE_ASYNC_LDS
typedef __attribute__((address_space(1))) v4i_g* gv4p;   // prints as __device__
typedef __attribute__((address_space(3))) v4i_g* lv4p;   // prints as __shared__
#endif

union BFrag { v16bf v; u32x4 q[2]; };
union F4 { float4 v; float a[4]; };

__device__ __forceinline__ unsigned short f2bfu(float f) {
  unsigned u = __builtin_bit_cast(unsigned, f);
  unsigned r = u + 0x7FFFu + ((u >> 16) & 1u);   // round-to-nearest-even
  return (unsigned short)(r >> 16);
}

__device__ __forceinline__ unsigned pack2bf(float lo, float hi) {
  return (unsigned)f2bfu(lo) | ((unsigned)f2bfu(hi) << 16);
}

// ==================================================================
// Kernel 0: zero output + per-expert counts
// ==================================================================
__global__ void moe_init_kernel(float* __restrict__ out, int* __restrict__ counts) {
  int i = blockIdx.x * blockDim.x + threadIdx.x;
  if (i < T_ * H_) out[i] = 0.0f;
  if (i < E_) counts[i] = 0;
}

// ==================================================================
// Kernel 1: RMSNorm + router (softmax/top-2) + gather lists
// one 256-thread block per token
// ==================================================================
__global__ void moe_router_kernel(const float* __restrict__ x,
                                  const float* __restrict__ nw,
                                  const float* __restrict__ rw,
                                  unsigned short* __restrict__ tbf,
                                  float* __restrict__ gates,
                                  int*   __restrict__ tok,
                                  int*   __restrict__ counts) {
  __shared__ float red[256];
  __shared__ float lg[E_];
  const int tid = threadIdx.x;
  const int t   = blockIdx.x;

  float xr[8];
  float ss = 0.f;
#pragma unroll
  for (int j = 0; j < 8; ++j) {
    int idx = j * 256 + tid;
    xr[j] = x[(size_t)t * H_ + idx];
    ss += xr[j] * xr[j];
  }
  red[tid] = ss; __syncthreads();
  for (int s = 128; s > 0; s >>= 1) {
    if (tid < s) red[tid] += red[tid + s];
    __syncthreads();
  }
  const float rinv = rsqrtf(red[0] * (1.0f / H_) + EPS_);
  __syncthreads();                         // red[] reused below

  float tn[8];
#pragma unroll
  for (int j = 0; j < 8; ++j) {
    int idx = j * 256 + tid;
    tn[j] = xr[j] * rinv * nw[idx];
    tbf[(size_t)t * H_ + idx] = f2bfu(tn[j]);
  }

  float pe[E_];
#pragma unroll
  for (int e = 0; e < E_; ++e) pe[e] = 0.f;
#pragma unroll
  for (int j = 0; j < 8; ++j) {
    int idx = j * 256 + tid;
#pragma unroll
    for (int e = 0; e < E_; ++e) pe[e] += tn[j] * rw[(size_t)e * H_ + idx];
  }
  for (int e = 0; e < E_; ++e) {
    red[tid] = pe[e]; __syncthreads();
    for (int s = 128; s > 0; s >>= 1) {
      if (tid < s) red[tid] += red[tid + s];
      __syncthreads();
    }
    if (tid == 0) lg[e] = red[0];
    __syncthreads();
  }

  if (tid == 0) {
    // top-2 of softmax(probs), renormalized == 2-way softmax over top-2 logits
    int   i0 = 0; float b0 = lg[0];
#pragma unroll
    for (int e = 1; e < E_; ++e) { if (lg[e] > b0) { b0 = lg[e]; i0 = e; } }
    int   i1 = -1; float b1 = -3.4e38f;
#pragma unroll
    for (int e = 0; e < E_; ++e) { if (e != i0 && lg[e] > b1) { b1 = lg[e]; i1 = e; } }
    float e1 = __expf(b1 - b0);
    float w0 = 1.0f / (1.0f + e1);
    float w1 = 1.0f - w0;
    int s0 = atomicAdd(&counts[i0], 1);
    tok[i0 * T_ + s0] = t; gates[i0 * T_ + s0] = w0;
    int s1 = atomicAdd(&counts[i1], 1);
    tok[i1 * T_ + s1] = t; gates[i1 * T_ + s1] = w1;
  }
}

// ==================================================================
// shared GEMM inner machinery
// Alds: [MTILE][KSTEP] bf16 (row-major, k contiguous)
// Blds: [NBLK][KSTEP]  bf16 (transposed: f-major, k contiguous)
// ==================================================================

// stage one 16-byte chunk of the A slab (64 rows x 32 k bf16 = 256 x 16B)
// src: bf16 row base (global), one chunk per thread
__device__ __forceinline__ void stage_a_chunk(unsigned short* Alds,
                                              const unsigned short* __restrict__ srcRow,
                                              int m, int ch) {
#ifdef HAVE_ASYNC_LDS
  __builtin_amdgcn_global_load_async_to_lds_b128(
      (gv4p)&srcRow[ch * 8],
      (lv4p)&Alds[m * KSTEP + ch * 8], 0, 0);
#else
  *(u32x4*)&Alds[m * KSTEP + ch * 8] = *(const u32x4*)&srcRow[ch * 8];
#endif
}

__device__ __forceinline__ void wait_async_then_barrier() {
#ifdef HAVE_ASYNC_LDS
  asm volatile("s_wait_asynccnt 0x0" ::: "memory");
#endif
  __syncthreads();
}

// fill B slab: KSTEP k-rows x NBLK f-cols of fp32 weights (row stride ldb)
// -> bf16 transposed in LDS, packed 2-per-b32 store
__device__ __forceinline__ void fill_b_slab(unsigned short* Blds,
                                            const float* __restrict__ wp,
                                            int k0, int ldb, int tid) {
  // work item: (kp, f4): k pair = 2*kp, 4 consecutive f
  // total = 16 kp * (NBLK/4=128) f4 = 2048 items, 8 per thread
#pragma unroll
  for (int j = 0; j < 8; ++j) {
    int i  = tid + j * 256;
    int kp = i >> 7;          // 0..15
    int f4 = i & 127;         // 0..127
    int k  = kp * 2;
    F4 r0, r1;
    r0.v = *(const float4*)&wp[(size_t)(k0 + k)     * ldb + f4 * 4];
    r1.v = *(const float4*)&wp[(size_t)(k0 + k + 1) * ldb + f4 * 4];
#pragma unroll
    for (int d = 0; d < 4; ++d) {
      int f = f4 * 4 + d;
      *(unsigned*)&Blds[f * KSTEP + k] = pack2bf(r0.a[d], r1.a[d]);
    }
  }
}

// load bf16 A fragment for 16x32 tile at row base m0 (ISA lane layout)
__device__ __forceinline__ BFrag load_a_frag(const unsigned short* Alds,
                                             int m0, int lrow, int half) {
  BFrag a;
  a.q[0] = *(const u32x4*)&Alds[(m0 + lrow) * KSTEP + half * 8];
  a.q[1] = *(const u32x4*)&Alds[(m0 + lrow) * KSTEP + half * 8 + 16];
  return a;
}

// load bf16 B fragment for 32x16 tile at col base f0 (ISA lane layout)
__device__ __forceinline__ BFrag load_b_frag(const unsigned short* Blds,
                                             int f0, int lrow, int half) {
  BFrag b;
  b.q[0] = *(const u32x4*)&Blds[(f0 + lrow) * KSTEP + half * 8];
  b.q[1] = *(const u32x4*)&Blds[(f0 + lrow) * KSTEP + half * 8 + 16];
  return b;
}

// ==================================================================
// Kernel 2: gathered GEMM1 (t @ w_gu[e]) + clamped SwiGLU -> act (bf16)
// grid = (TWO_I/NBLK, T/MTILE, E), block = 256
// ==================================================================
__global__ void __launch_bounds__(256)
moe_gemm1_swiglu_kernel(const unsigned short* __restrict__ tbf,
                        const float* __restrict__ wgu,
                        const int*   __restrict__ tok,
                        const int*   __restrict__ counts,
                        unsigned short* __restrict__ act) {
  const int e  = blockIdx.z;
  const int mt = blockIdx.y;
  const int n  = counts[e];
  if (mt * MTILE >= n) return;
  const int nbase = blockIdx.x * NBLK;

  __shared__ __align__(16) unsigned short Alds[MTILE * KSTEP];  // 4 KB
  __shared__ __align__(16) unsigned short Blds[NBLK * KSTEP];   // 32 KB
  __shared__ int stok[MTILE];

  const int tid  = threadIdx.x;
  const int lane = tid & 31;
  const int wave = tid >> 5;
  const int half = lane >> 4;
  const int lrow = lane & 15;
  const int am   = tid >> 2;   // A-slab row handled by this thread
  const int ach  = tid & 3;    // 16B chunk within the 64B k-slab row

  if (tid < MTILE) {
    int r = mt * MTILE + tid;
    stok[tid] = (r < n) ? tok[e * T_ + r] : tok[e * T_];   // safe fallback row
  }
  __syncthreads();

  const float* wp = wgu + (size_t)e * H_ * TWO_I + nbase;
  const unsigned short* arow = tbf + (size_t)stok[am] * H_;

  v8f c[4][4];
#pragma unroll
  for (int mi = 0; mi < 4; ++mi)
#pragma unroll
    for (int ni = 0; ni < 4; ++ni) c[mi][ni] = {};

  for (int kt = 0; kt < KT_H; ++kt) {
    const int k0 = kt * KSTEP;
    stage_a_chunk(Alds, arow + k0, am, ach);     // async DMA when available
    fill_b_slab(Blds, wp, k0, TWO_I, tid);
    if (kt + 1 < KT_H)
      __builtin_prefetch(&wp[(size_t)(k0 + KSTEP + (tid >> 4)) * TWO_I + (tid & 15) * 32], 0, 1);
    wait_async_then_barrier();

    BFrag a[4];
#pragma unroll
    for (int mi = 0; mi < 4; ++mi) a[mi] = load_a_frag(Alds, mi * 16, lrow, half);
#pragma unroll
    for (int ni = 0; ni < 4; ++ni) {
      BFrag b = load_b_frag(Blds, wave * 64 + ni * 16, lrow, half);
#pragma unroll
      for (int mi = 0; mi < 4; ++mi)
        c[mi][ni] = __builtin_amdgcn_wmma_f32_16x16x32_bf16(
            false, a[mi].v, false, b.v, (short)0, c[mi][ni], false, false);
    }
    __syncthreads();
  }

  // SwiGLU epilogue: interleaved gate/linear columns are adjacent lanes
#pragma unroll
  for (int mi = 0; mi < 4; ++mi)
#pragma unroll
    for (int ni = 0; ni < 4; ++ni)
#pragma unroll
      for (int g = 0; g < 8; ++g) {
        float v     = c[mi][ni][g];
        float other = __shfl_xor(v, 1, 32);
        if ((lane & 1) == 0) {
          float glu = fminf(v, LIMIT_);
          float lin = fminf(fmaxf(other, -LIMIT_), LIMIT_);
          float av  = glu * (1.0f / (1.0f + __expf(-ALPHA_ * glu))) * (lin + 1.0f);
          int m = mi * 16 + g + 8 * half;
          int r = mt * MTILE + m;
          if (r < n) {
            int f = nbase + wave * 64 + ni * 16 + lrow;   // even
            act[((size_t)e * T_ + r) * I_ + (f >> 1)] = f2bfu(av);
          }
        }
      }
}

// ==================================================================
// Kernel 3: gathered GEMM2 (act @ w_down[e]) * gate -> atomic scatter-add
// grid = (H/NBLK, T/MTILE, E), block = 256
// ==================================================================
__global__ void __launch_bounds__(256)
moe_gemm2_scatter_kernel(const unsigned short* __restrict__ act,
                         const float* __restrict__ wd,
                         const int*   __restrict__ tok,
                         const float* __restrict__ gates,
                         const int*   __restrict__ counts,
                         float* __restrict__ out) {
  const int e  = blockIdx.z;
  const int mt = blockIdx.y;
  const int n  = counts[e];
  if (mt * MTILE >= n) return;
  const int nbase = blockIdx.x * NBLK;

  __shared__ __align__(16) unsigned short Alds[MTILE * KSTEP];
  __shared__ __align__(16) unsigned short Blds[NBLK * KSTEP];
  __shared__ int   stok[MTILE];
  __shared__ float sgate[MTILE];

  const int tid  = threadIdx.x;
  const int lane = tid & 31;
  const int wave = tid >> 5;
  const int half = lane >> 4;
  const int lrow = lane & 15;
  const int am   = tid >> 2;
  const int ach  = tid & 3;

  if (tid < MTILE) {
    int r = mt * MTILE + tid;
    stok[tid]  = (r < n) ? tok[e * T_ + r]   : tok[e * T_];
    sgate[tid] = (r < n) ? gates[e * T_ + r] : 0.0f;
  }
  __syncthreads();

  const float* wp = wd + (size_t)e * I_ * H_ + nbase;
  const unsigned short* ap = act + ((size_t)e * T_ + mt * MTILE) * I_;
  const unsigned short* arow = ap + (size_t)am * I_;

  v8f c[4][4];
#pragma unroll
  for (int mi = 0; mi < 4; ++mi)
#pragma unroll
    for (int ni = 0; ni < 4; ++ni) c[mi][ni] = {};

  for (int kt = 0; kt < KT_I; ++kt) {
    const int k0 = kt * KSTEP;
    stage_a_chunk(Alds, arow + k0, am, ach);
    fill_b_slab(Blds, wp, k0, H_, tid);
    if (kt + 1 < KT_I)
      __builtin_prefetch(&wp[(size_t)(k0 + KSTEP + (tid >> 4)) * H_ + (tid & 15) * 32], 0, 1);
    wait_async_then_barrier();

    BFrag a[4];
#pragma unroll
    for (int mi = 0; mi < 4; ++mi) a[mi] = load_a_frag(Alds, mi * 16, lrow, half);
#pragma unroll
    for (int ni = 0; ni < 4; ++ni) {
      BFrag b = load_b_frag(Blds, wave * 64 + ni * 16, lrow, half);
#pragma unroll
      for (int mi = 0; mi < 4; ++mi)
        c[mi][ni] = __builtin_amdgcn_wmma_f32_16x16x32_bf16(
            false, a[mi].v, false, b.v, (short)0, c[mi][ni], false, false);
    }
    __syncthreads();
  }

#pragma unroll
  for (int mi = 0; mi < 4; ++mi)
#pragma unroll
    for (int ni = 0; ni < 4; ++ni)
#pragma unroll
      for (int g = 0; g < 8; ++g) {
        int m = mi * 16 + g + 8 * half;
        int r = mt * MTILE + m;
        if (r < n) {
          int h = nbase + wave * 64 + ni * 16 + lrow;
          atomicAdd(&out[(size_t)stok[m] * H_ + h], c[mi][ni][g] * sgate[m]);
        }
      }
}

// ==================================================================
// host-side launch
// ==================================================================
extern "C" void kernel_launch(void* const* d_in, const int* in_sizes, int n_in,
                              void* d_out, int out_size, void* d_ws, size_t ws_size,
                              hipStream_t stream) {
  (void)in_sizes; (void)n_in; (void)out_size; (void)ws_size;
  const float* x        = (const float*)d_in[0];
  const float* norm_w   = (const float*)d_in[1];
  const float* router_w = (const float*)d_in[2];
  const float* w_gu     = (const float*)d_in[3];
  const float* w_down   = (const float*)d_in[4];
  // d_in[5] = top_k (== 2, hardcoded as TOPK_)
  float* out = (float*)d_out;

  unsigned char* ws = (unsigned char*)d_ws;
  unsigned short* tbf   = (unsigned short*)(ws + OFF_TBF);
  float*          gates = (float*)(ws + OFF_GATES);
  int*            tok   = (int*)(ws + OFF_TOK);
  int*            cnt   = (int*)(ws + OFF_CNT);
  unsigned short* act   = (unsigned short*)(ws + OFF_ACT);

  moe_init_kernel<<<(T_ * H_ + 255) / 256, 256, 0, stream>>>(out, cnt);
  moe_router_kernel<<<T_, 256, 0, stream>>>(x, norm_w, router_w, tbf, gates, tok, cnt);
  moe_gemm1_swiglu_kernel<<<dim3(TWO_I / NBLK, T_ / MTILE, E_), 256, 0, stream>>>(
      tbf, w_gu, tok, cnt, act);
  moe_gemm2_scatter_kernel<<<dim3(H_ / NBLK, T_ / MTILE, E_), 256, 0, stream>>>(
      act, w_down, tok, gates, cnt, out);
}
